// TransformerEncoderLayer_69243462746781
// MI455X (gfx1250) — compile-verified, matrix operand order checked by reading, not run on hardware
//
#include <hip/hip_runtime.h>
#include <hip/hip_bf16.h>
#include <cstdint>

typedef __bf16 bf16;
typedef __attribute__((ext_vector_type(16))) __bf16 v16bf;
typedef __attribute__((ext_vector_type(8)))  __bf16 v8bf;
typedef __attribute__((ext_vector_type(4)))  __bf16 v4bf;
typedef __attribute__((ext_vector_type(8)))  float  v8f;

#define B_   2
#define N_   2048
#define E_   1024
#define H_   16
#define D_   64
#define MLP_ 4096
#define TOK  (B_*N_)      // 4096 tokens
#define LDT  40           // padded LDS row stride (elems): 32 bf16 data + 8 pad = 80B

// ---------------------------------------------------------------------------
// CDNA5 helpers
// ---------------------------------------------------------------------------
__device__ __forceinline__ v8f wmma_bf16(v16bf a, v16bf b, v8f c) {
  // D = A(16x32 bf16) * B(32x16 bf16) + C(16x16 f32)
  return __builtin_amdgcn_wmma_f32_16x16x32_bf16(false, a, false, b,
                                                 (short)0, c, false, false);
}

__device__ __forceinline__ uint32_t lds_off(const void* p) {
  // generic pointer to LDS: low 32 bits are the LDS byte offset (aperture in high bits)
  return (uint32_t)(uintptr_t)p;
}

__device__ __forceinline__ void async_copy_b128(uint32_t ldsaddr, const void* g) {
  asm volatile("global_load_async_to_lds_b128 %0, %1, off"
               :: "v"(ldsaddr), "v"((uint64_t)(uintptr_t)g) : "memory");
}
__device__ __forceinline__ void wait_async0() {
  asm volatile("s_wait_asynccnt 0" ::: "memory");
}

// Load one 16x32 bf16 WMMA fragment (A-layout; B-operand uses same pattern over
// a [N,K] row-major "Bt" buffer). lane r = lane&15 -> row, kh = lane>>4 selects
// which two contiguous 8-elem K chunks this lane holds (ISA 7.12.2).
__device__ __forceinline__ v16bf frag_ld(const bf16* base, int rstride, int lane) {
  const int r = lane & 15, kh = lane >> 4;
  const v8bf* p0 = (const v8bf*)(base + (size_t)r * rstride + kh * 8);
  const v8bf* p1 = (const v8bf*)(base + (size_t)r * rstride + kh * 8 + 16);
  v8bf lo = *p0, hi = *p1;
  return __builtin_shufflevector(lo, hi, 0,1,2,3,4,5,6,7,8,9,10,11,12,13,14,15);
}

// ---------------------------------------------------------------------------
// LayerNorm (fp32 in -> bf16 out), one 256-thread block per 1024-elem row
// ---------------------------------------------------------------------------
__global__ __launch_bounds__(256)
void ln_kernel(const float* __restrict__ x, const float* __restrict__ g,
               const float* __restrict__ be, bf16* __restrict__ out)
{
  const int row = blockIdx.x;
  const int tid = threadIdx.x;
  const float4 v = ((const float4*)(x + (size_t)row * E_))[tid];
  float s  = v.x + v.y + v.z + v.w;
  float sq = v.x*v.x + v.y*v.y + v.z*v.z + v.w*v.w;
  #pragma unroll
  for (int off = 16; off >= 1; off >>= 1) {
    s  += __shfl_xor(s,  off, 32);
    sq += __shfl_xor(sq, off, 32);
  }
  __shared__ float rs[8], rq[8];
  const int wave = tid >> 5, lane = tid & 31;
  if (lane == 0) { rs[wave] = s; rq[wave] = sq; }
  __syncthreads();
  float ts = 0.f, tq = 0.f;
  #pragma unroll
  for (int i = 0; i < 8; ++i) { ts += rs[i]; tq += rq[i]; }
  const float mu  = ts * (1.0f / E_);
  const float var = tq * (1.0f / E_) - mu * mu;
  const float inv = rsqrtf(var + 1e-5f);
  const int c = tid * 4;
  v4bf o;
  o[0] = (bf16)((v.x - mu) * inv * g[c+0] + be[c+0]);
  o[1] = (bf16)((v.y - mu) * inv * g[c+1] + be[c+1]);
  o[2] = (bf16)((v.z - mu) * inv * g[c+2] + be[c+2]);
  o[3] = (bf16)((v.w - mu) * inv * g[c+3] + be[c+3]);
  *(v4bf*)(out + (size_t)row * E_ + c) = o;
}

// ---------------------------------------------------------------------------
// Transpose + fp32->bf16 convert: in [K, Nout] fp32 -> out [Nout, K] bf16
// ---------------------------------------------------------------------------
__global__ __launch_bounds__(256)
void transpose_conv_kernel(const float* __restrict__ in, bf16* __restrict__ out,
                           int K, int Nout)
{
  __shared__ float t[32][33];
  const int tx = threadIdx.x & 31, ty = threadIdx.x >> 5;
  const int n0 = blockIdx.x * 32, k0 = blockIdx.y * 32;
  #pragma unroll
  for (int i = 0; i < 4; ++i)
    t[ty + i*8][tx] = in[(size_t)(k0 + ty + i*8) * Nout + n0 + tx];
  __syncthreads();
  #pragma unroll
  for (int i = 0; i < 4; ++i)
    out[(size_t)(n0 + ty + i*8) * K + k0 + tx] = (bf16)t[tx][ty + i*8];
}

// ---------------------------------------------------------------------------
// bf16 WMMA GEMM: C[M,Nout] = A[M,K] * Bt[Nout,K]^T + bias, with epilogues.
// Block = 256 thr (8 waves, 2x4 grid), WG tile 128x256, wave tile 64x64,
// K-step 32, double-buffered async global->LDS staging.
// Per wave per K-step: 16 WMMAs vs 8 fragments (16 ds_load_b128) = 1.0 ld/wmma.
// MODE 1: qkv -> bf16 out, V-part additionally scattered into vT [B,H,D,N]
// MODE 2: fp32 out = res + acc + bias   (residual add)
// MODE 3: bf16 out = gelu(acc + bias)   (exact gelu, erf)
// ---------------------------------------------------------------------------
template<int MODE>
__global__ __launch_bounds__(256)
void gemm_bf16_kernel(const bf16* __restrict__ A, const bf16* __restrict__ Bt,
                      const float* __restrict__ bias, const float* __restrict__ res,
                      float* __restrict__ outf, bf16* __restrict__ outb,
                      bf16* __restrict__ vT, int M, int Nout, int K)
{
  __shared__ bf16 sA[2][128 * LDT];   // 20 KB
  __shared__ bf16 sB[2][256 * LDT];   // 40 KB
  const int tid  = threadIdx.x;
  const int lane = tid & 31, wave = tid >> 5;
  const int wm = wave >> 2, wn = wave & 3;          // 2 x 4 wave grid
  const int mbase = blockIdx.y * 128;
  const int nbase = blockIdx.x * 256;

  auto stage = [&](int buf, int kpos) {
    #pragma unroll
    for (int i = 0; i < 2; ++i) {                   // A: 128 rows x 4 chunks = 512
      const int c = tid + i * 256;
      const int row = c >> 2, kc = c & 3;
      async_copy_b128(lds_off(&sA[buf][row * LDT + kc * 8]),
                      A + (size_t)(mbase + row) * K + kpos + kc * 8);
    }
    #pragma unroll
    for (int i = 0; i < 4; ++i) {                   // B: 256 rows x 4 chunks = 1024
      const int c = tid + i * 256;
      const int row = c >> 2, kc = c & 3;
      async_copy_b128(lds_off(&sB[buf][row * LDT + kc * 8]),
                      Bt + (size_t)(nbase + row) * K + kpos + kc * 8);
    }
  };

  v8f acc[4][4];
  #pragma unroll
  for (int mt = 0; mt < 4; ++mt)
    #pragma unroll
    for (int nt = 0; nt < 4; ++nt)
      acc[mt][nt] = (v8f){0.f,0.f,0.f,0.f,0.f,0.f,0.f,0.f};

  const int steps = K / 32;
  stage(0, 0);
  for (int s = 0; s < steps; ++s) {
    const int buf = s & 1;
    wait_async0();            // this wave's async copies for buf are in LDS
    __syncthreads();          // every wave's copies are in LDS; prev reads done
    if (s + 1 < steps) stage(buf ^ 1, (s + 1) * 32);

    const bf16* pA = &sA[buf][wm * 64 * LDT];
    const bf16* pB = &sB[buf][wn * 64 * LDT];
    v16bf af[4];
    #pragma unroll
    for (int mt = 0; mt < 4; ++mt) af[mt] = frag_ld(pA + mt * 16 * LDT, LDT, lane);
    #pragma unroll
    for (int nt = 0; nt < 4; ++nt) {
      const v16bf bfr = frag_ld(pB + nt * 16 * LDT, LDT, lane);
      #pragma unroll
      for (int mt = 0; mt < 4; ++mt)
        acc[mt][nt] = wmma_bf16(af[mt], bfr, acc[mt][nt]);
    }
  }

  // epilogue: C layout — lane holds (row = (lane>>4)*8+j, col = lane&15) per tile
  #pragma unroll
  for (int mt = 0; mt < 4; ++mt) {
    #pragma unroll
    for (int nt = 0; nt < 4; ++nt) {
      #pragma unroll
      for (int j = 0; j < 8; ++j) {
        const int row = mbase + wm * 64 + mt * 16 + (lane >> 4) * 8 + j;
        const int col = nbase + wn * 64 + nt * 16 + (lane & 15);
        const float v = acc[mt][nt][j] + bias[col];
        if (MODE == 1) {
          outb[(size_t)row * Nout + col] = (bf16)v;
          if (col >= 2 * E_) {                      // V part -> transposed copy
            const int dd = col - 2 * E_;
            const int h = dd >> 6, d = dd & 63;
            const int b = row >> 11, n = row & (N_ - 1);
            vT[((size_t)((b * H_ + h) * D_ + d)) * N_ + n] = (bf16)v;
          }
        } else if (MODE == 2) {
          outf[(size_t)row * Nout + col] = res[(size_t)row * Nout + col] + v;
        } else {
          const float ge = 0.5f * v * (1.0f + erff(v * 0.70710678118654752f));
          outb[(size_t)row * Nout + col] = (bf16)ge;
        }
      }
    }
  }
}

// ---------------------------------------------------------------------------
// Flash attention: one 256-thr block = 8 waves, each wave owns 16 query rows.
// grid = B*H*(N/128). qkv bf16 [TOK,3E] (q|k|v), vT bf16 [B,H,D,N].
// Online softmax, S=Q*K^T and O+=P*V on bf16 WMMA. out bf16 [TOK, E].
// ---------------------------------------------------------------------------
__global__ __launch_bounds__(256)
void attention_kernel(const bf16* __restrict__ qkv, const bf16* __restrict__ vT,
                      bf16* __restrict__ out)
{
  __shared__ bf16 sP[8][16 * LDT];
  const int tid = threadIdx.x, lane = tid & 31, wave = tid >> 5;
  const int wg = blockIdx.x;
  const int rowblk = wg & 15;          // N/128 = 16 row blocks
  const int bh = wg >> 4;
  const int b = bh >> 4, h = bh & 15;
  const int q0 = rowblk * 128 + wave * 16;
  const size_t tok0 = (size_t)b * N_ + q0;

  const bf16* Q  = qkv + tok0 * (3 * E_) + h * D_;
  const bf16* Kp = qkv + (size_t)b * N_ * (3 * E_) + E_ + h * D_;
  const bf16* Vh = vT  + (size_t)((b * H_ + h) * D_) * N_;

  const v16bf qa0 = frag_ld(Q,      3 * E_, lane);   // d 0..31
  const v16bf qa1 = frag_ld(Q + 32, 3 * E_, lane);   // d 32..63

  v8f o[4];
  float mrun[8], lrun[8];
  #pragma unroll
  for (int t = 0; t < 4; ++t) o[t] = (v8f){0,0,0,0,0,0,0,0};
  #pragma unroll
  for (int j = 0; j < 8; ++j) { mrun[j] = -1e30f; lrun[j] = 0.f; }

  const float scale = 0.125f;          // 1/sqrt(64)
  bf16* pbuf = &sP[wave][0];

  for (int kb = 0; kb < N_; kb += 32) {
    // ---- S = scale * Q K^T for 32 keys (two 16x16 tiles) ----
    const v16bf k00 = frag_ld(Kp + (size_t)kb       * (3*E_),      3*E_, lane);
    const v16bf k01 = frag_ld(Kp + (size_t)kb       * (3*E_) + 32, 3*E_, lane);
    const v16bf k10 = frag_ld(Kp + (size_t)(kb+16)  * (3*E_),      3*E_, lane);
    const v16bf k11 = frag_ld(Kp + (size_t)(kb+16)  * (3*E_) + 32, 3*E_, lane);
    const v8f z = (v8f){0,0,0,0,0,0,0,0};
    v8f s0 = wmma_bf16(qa1, k01, wmma_bf16(qa0, k00, z));
    v8f s1 = wmma_bf16(qa1, k11, wmma_bf16(qa0, k10, z));

    // ---- online softmax over this 32-key block ----
    float mt[8];
    #pragma unroll
    for (int j = 0; j < 8; ++j) {
      s0[j] *= scale; s1[j] *= scale;
      mt[j] = fmaxf(s0[j], s1[j]);
    }
    #pragma unroll
    for (int off = 1; off < 16; off <<= 1)
      #pragma unroll
      for (int j = 0; j < 8; ++j)
        mt[j] = fmaxf(mt[j], __shfl_xor(mt[j], off, 32));  // reduce 16-lane row group

    float corr[8], p0[8], p1[8], rt[8];
    #pragma unroll
    for (int j = 0; j < 8; ++j) {
      const float mn = fmaxf(mrun[j], mt[j]);
      corr[j] = __expf(mrun[j] - mn);
      p0[j] = __expf(s0[j] - mn);
      p1[j] = __expf(s1[j] - mn);
      mrun[j] = mn;
      rt[j] = p0[j] + p1[j];
    }
    #pragma unroll
    for (int off = 1; off < 16; off <<= 1)
      #pragma unroll
      for (int j = 0; j < 8; ++j)
        rt[j] += __shfl_xor(rt[j], off, 32);
    #pragma unroll
    for (int j = 0; j < 8; ++j) lrun[j] = lrun[j] * corr[j] + rt[j];
    #pragma unroll
    for (int t = 0; t < 4; ++t)
      #pragma unroll
      for (int j = 0; j < 8; ++j) o[t][j] *= corr[j];

    // ---- C-layout -> A-layout via wave-private LDS bounce ----
    const int rb = (lane >> 4) * 8, cc = lane & 15;
    #pragma unroll
    for (int j = 0; j < 8; ++j) {
      pbuf[(rb + j) * LDT + cc]      = (bf16)p0[j];
      pbuf[(rb + j) * LDT + 16 + cc] = (bf16)p1[j];
    }
    const v16bf pa = frag_ld((const bf16*)pbuf, LDT, lane);

    // ---- O += P * V (V pre-transposed: rows = d, contiguous tokens) ----
    #pragma unroll
    for (int t = 0; t < 4; ++t) {
      const v16bf vf = frag_ld(Vh + (size_t)(t * 16) * N_ + kb, N_, lane);
      o[t] = wmma_bf16(pa, vf, o[t]);
    }
  }

  // ---- normalize and store bf16 [TOK, E] ----
  #pragma unroll
  for (int j = 0; j < 8; ++j) {
    const float inv = 1.0f / lrun[j];
    const int r = (lane >> 4) * 8 + j;
    const size_t orow = (tok0 + r) * E_;
    #pragma unroll
    for (int t = 0; t < 4; ++t)
      out[orow + h * D_ + t * 16 + (lane & 15)] = (bf16)(o[t][j] * inv);
  }
}

// ---------------------------------------------------------------------------
// Host-side orchestration
// ---------------------------------------------------------------------------
extern "C" void kernel_launch(void* const* d_in, const int* in_sizes, int n_in,
                              void* d_out, int out_size, void* d_ws, size_t ws_size,
                              hipStream_t stream)
{
  (void)in_sizes; (void)n_in; (void)out_size; (void)ws_size;
  const float* x      = (const float*)d_in[0];
  const float* w_qkv  = (const float*)d_in[1];
  const float* b_qkv  = (const float*)d_in[2];
  const float* w_proj = (const float*)d_in[3];
  const float* b_proj = (const float*)d_in[4];
  const float* g1     = (const float*)d_in[5];
  const float* beta1  = (const float*)d_in[6];
  const float* g2     = (const float*)d_in[7];
  const float* beta2  = (const float*)d_in[8];
  const float* w_fc1  = (const float*)d_in[9];
  const float* b_fc1  = (const float*)d_in[10];
  const float* w_fc2  = (const float*)d_in[11];
  const float* b_fc2  = (const float*)d_in[12];

  char* ws = (char*)d_ws;
  const size_t MB = 1ull << 20;
  // lifetime-overlapped workspace (64 MB total):
  bf16* wT   = (bf16*)(ws + 0);          //  8 MB: transposed weight scratch (reused 4x)
  bf16* regA = (bf16*)(ws + 8 * MB);     // 32 MB: qkv (24MB) then gelu (32MB)
  char* regB = ws + 40 * MB;             // 16 MB: vT then x2
  bf16* regC = (bf16*)(ws + 56 * MB);    //  8 MB: h1 -> attn_out -> h2

  bf16*  h1       = regC;
  bf16*  qkv      = regA;
  bf16*  vT       = (bf16*)regB;
  bf16*  attn_out = regC;
  float* x2       = (float*)regB;
  bf16*  h2       = regC;
  bf16*  gel      = regA;
  float* outf     = (float*)d_out;

  const dim3 blk(256);

  // 1) h1 = LN1(x)
  ln_kernel<<<TOK, blk, 0, stream>>>(x, g1, beta1, h1);
  // 2) wT = (w_qkv)^T bf16  [3E, E]
  transpose_conv_kernel<<<dim3(3*E_/32, E_/32), blk, 0, stream>>>(w_qkv, wT, E_, 3*E_);
  // 3) qkv = h1 @ w_qkv + b_qkv (bf16), V scattered into vT
  gemm_bf16_kernel<1><<<dim3(3*E_/256, TOK/128), blk, 0, stream>>>(
      h1, wT, b_qkv, nullptr, nullptr, qkv, vT, TOK, 3*E_, E_);
  // 4) attn_out = softmax(QK^T/sqrt(d)) V  (flash, bf16)
  attention_kernel<<<dim3(B_*H_*(N_/128)), blk, 0, stream>>>(qkv, vT, attn_out);
  // 5) wT = (w_proj)^T
  transpose_conv_kernel<<<dim3(E_/32, E_/32), blk, 0, stream>>>(w_proj, wT, E_, E_);
  // 6) x2 = x + attn_out @ w_proj + b_proj  (fp32)
  gemm_bf16_kernel<2><<<dim3(E_/256, TOK/128), blk, 0, stream>>>(
      attn_out, wT, b_proj, x, x2, nullptr, nullptr, TOK, E_, E_);
  // 7) h2 = LN2(x2)
  ln_kernel<<<TOK, blk, 0, stream>>>(x2, g2, beta2, h2);
  // 8) wT = (w_fc1)^T  [MLP, E]
  transpose_conv_kernel<<<dim3(MLP_/32, E_/32), blk, 0, stream>>>(w_fc1, wT, E_, MLP_);
  // 9) gel = gelu(h2 @ w_fc1 + b_fc1) (bf16)
  gemm_bf16_kernel<3><<<dim3(MLP_/256, TOK/128), blk, 0, stream>>>(
      h2, wT, b_fc1, nullptr, nullptr, gel, nullptr, TOK, MLP_, E_);
  // 10) wT = (w_fc2)^T  [E, MLP]
  transpose_conv_kernel<<<dim3(E_/32, MLP_/32), blk, 0, stream>>>(w_fc2, wT, MLP_, E_);
  // 11) out = x2 + gel @ w_fc2 + b_fc2  (fp32)
  gemm_bf16_kernel<2><<<dim3(E_/256, TOK/128), blk, 0, stream>>>(
      gel, wT, b_fc2, x2, outf, nullptr, nullptr, TOK, E_, MLP_);
}
